// VectorQuantizer_27625229648009
// MI455X (gfx1250) — matrix operand order repaired and checked by reading, not compiled, not run on hardware
//
#include <hip/hip_runtime.h>
#include <hip/hip_bf16.h>

typedef __bf16 bf16_t;
typedef __attribute__((ext_vector_type(16))) __bf16 v16bf;
typedef __attribute__((ext_vector_type(8)))  float  v8f;

#define N_ROWS 65536
#define DIM    256
#define KCODES 8192
#define KTILES (KCODES / 16)      // 512
#define ROWS_PER_BLOCK 128
#define NBLOCKS (N_ROWS / ROWS_PER_BLOCK)   // 512

static __device__ __forceinline__ __bf16 f2bf(float x) {
    unsigned u = __builtin_bit_cast(unsigned, x);
    unsigned r = (u + 0x7FFFu + ((u >> 16) & 1u)) >> 16;   // round-nearest-even
    unsigned short hs = (unsigned short)r;
    return __builtin_bit_cast(__bf16, hs);
}
static __device__ __forceinline__ float bf2f(__bf16 h) {
    unsigned short hs = __builtin_bit_cast(unsigned short, h);
    unsigned u = ((unsigned)hs) << 16;
    return __builtin_bit_cast(float, u);
}

// 128-bit async global->LDS copy (register-free, tracked by ASYNCcnt).
static __device__ __forceinline__ void async_copy_b128(unsigned lds_byte_addr,
                                                       const void* gaddr) {
    asm volatile("global_load_async_to_lds_b128 %0, %1, off"
                 :
                 : "v"(lds_byte_addr), "v"((unsigned long long)(uintptr_t)gaddr)
                 : "memory");
}
static __device__ __forceinline__ void wait_asynccnt0() {
    asm volatile("s_wait_asynccnt 0x0" ::: "memory");
}

// ---------------------------------------------------------------------------
// Pre-swizzle W into CDNA5 B-fragment layout (hi/lo bf16 split).
// B operand order used here (contiguous k, as A): lane l -> code n = l&15, K-half = l>>4,
// element e (0..15) -> K = khalf*16 + e. Stored per (tile t, kstep s):
// dst[((t*8+s)*32 + l)*16 + e]
// ---------------------------------------------------------------------------
__global__ __launch_bounds__(256) void vq_prep_codes(const float* __restrict__ W,
                                                     bf16_t* __restrict__ whi,
                                                     bf16_t* __restrict__ wlo) {
    size_t o = (size_t)blockIdx.x * 256 + threadIdx.x;   // 0 .. 2M-1
    int e = (int)(o & 15);
    int l = (int)((o >> 4) & 31);
    int s = (int)((o >> 9) & 7);
    int t = (int)(o >> 12);
    int row = t * 16 + (l & 15);
    int k   = s * 32 + (l >> 4) * 16 + e;
    float x = W[(size_t)row * DIM + k];
    __bf16 hi = f2bf(x);
    whi[o] = hi;
    wlo[o] = f2bf(x - bf2f(hi));
}

// ||e_k||^2 in exact f32, one block per code.
__global__ __launch_bounds__(256) void vq_esq(const float* __restrict__ W,
                                              float* __restrict__ esq) {
    __shared__ float sred[256];
    int b = blockIdx.x;
    float x = W[(size_t)b * DIM + threadIdx.x];
    sred[threadIdx.x] = x * x;
    __syncthreads();
    for (int st = 128; st > 0; st >>= 1) {
        if ((int)threadIdx.x < st) sred[threadIdx.x] += sred[threadIdx.x + st];
        __syncthreads();
    }
    if (threadIdx.x == 0) esq[b] = sred[0];
}

// ---------------------------------------------------------------------------
// Main kernel: 8 waves/block, each wave owns 16 rows (A hi/lo resident in
// VGPRs), all waves sweep the full codebook sharing B tiles via LDS
// double-buffer filled with GLOBAL_LOAD_ASYNC_TO_LDS_B128 (ASYNCcnt).
// dist = ||e||^2 - 2 z.e (||z||^2 constant per row, dropped for argmin).
// ---------------------------------------------------------------------------
__global__ __launch_bounds__(256) void vq_main(const float* __restrict__ z,
                                               const float* __restrict__ W,
                                               const bf16_t* __restrict__ whi,
                                               const bf16_t* __restrict__ wlo,
                                               const float* __restrict__ esq,
                                               float* __restrict__ out_zq,
                                               float* __restrict__ out_idx,
                                               float* __restrict__ partials) {
    __shared__ __align__(32) bf16_t sB[2][2][8 * 32 * 16];  // [buf][hi/lo][4096] = 32KB
    __shared__ int   sIdx[ROWS_PER_BLOCK];
    __shared__ float sRed[256];

    const int tid  = threadIdx.x;
    const int wave = tid >> 5;
    const int lane = tid & 31;
    const int h    = lane >> 4;
    const int rowBase = blockIdx.x * ROWS_PER_BLOCK;
    const int mrow = rowBase + wave * 16 + (lane & 15);

    // ---- Build A fragments (16x32 bf16 per k-step), hi + lo split ----
    // lane l: M = l&15; K = s*32 + 16h + e
    v16bf Ahi[8], Alo[8];
    const float* zrow = z + (size_t)mrow * DIM;
#pragma unroll
    for (int s = 0; s < 8; ++s) {
#pragma unroll
        for (int e = 0; e < 16; ++e) {
            int k = s * 32 + 16 * h + e;
            float x = zrow[k];
            __bf16 hi = f2bf(x);
            Ahi[s][e] = hi;
            Alo[s][e] = f2bf(x - bf2f(hi));
        }
    }

    // LDS byte addresses for the ping-pong buffers (per-thread 16B chunk).
    // Each thread copies 2x16B of hi and 2x16B of lo per tile (8KB each total).
    const unsigned ldsH0 = (unsigned)(uintptr_t)(void*)&sB[0][0][0] + (unsigned)tid * 16u;
    const unsigned ldsL0 = (unsigned)(uintptr_t)(void*)&sB[0][1][0] + (unsigned)tid * 16u;
    const unsigned bufStride = (unsigned)((uintptr_t)(void*)&sB[1][0][0] -
                                          (uintptr_t)(void*)&sB[0][0][0]);

    auto stage_async = [&](int buf, int t) {
        const char* gH = (const char*)(whi + (size_t)t * 4096) + (size_t)tid * 16u;
        const char* gL = (const char*)(wlo + (size_t)t * 4096) + (size_t)tid * 16u;
        unsigned dH = ldsH0 + (unsigned)buf * bufStride;
        unsigned dL = ldsL0 + (unsigned)buf * bufStride;
        async_copy_b128(dH,          gH);
        async_copy_b128(dH + 4096u,  gH + 4096u);
        async_copy_b128(dL,          gL);
        async_copy_b128(dL + 4096u,  gL + 4096u);
    };

    float minv[8];
    int   mini[8];
#pragma unroll
    for (int j = 0; j < 8; ++j) { minv[j] = 3.4e38f; mini[j] = 0; }

    stage_async(0, 0);
    wait_asynccnt0();
    __syncthreads();

    for (int t = 0; t < KTILES; ++t) {
        const int buf = t & 1;
        if (t + 1 < KTILES) stage_async(buf ^ 1, t + 1);  // overlaps with WMMA below

        const v16bf* Bh = (const v16bf*)sB[buf][0];
        const v16bf* Bl = (const v16bf*)sB[buf][1];

        // Three independent accumulator chains for WMMA ILP; exact sum of the
        // three split-precision products is formed in the f32 epilogue.
        v8f accHH = {0.f, 0.f, 0.f, 0.f, 0.f, 0.f, 0.f, 0.f};
        v8f accHL = {0.f, 0.f, 0.f, 0.f, 0.f, 0.f, 0.f, 0.f};
        v8f accLH = {0.f, 0.f, 0.f, 0.f, 0.f, 0.f, 0.f, 0.f};
#pragma unroll
        for (int s = 0; s < 8; ++s) {
            v16bf bh = Bh[s * 32 + lane];
            v16bf bl = Bl[s * 32 + lane];
            accHH = __builtin_amdgcn_wmma_f32_16x16x32_bf16(false, Ahi[s], false, bh,
                                                            (short)0, accHH, false, false);
            accHL = __builtin_amdgcn_wmma_f32_16x16x32_bf16(false, Ahi[s], false, bl,
                                                            (short)0, accHL, false, false);
            accLH = __builtin_amdgcn_wmma_f32_16x16x32_bf16(false, Alo[s], false, bh,
                                                            (short)0, accLH, false, false);
        }

        // C layout: lane<16 -> row=j, col=lane ; lane>=16 -> row=j+8, col=lane-16
        const int n  = t * 16 + (lane & 15);
        const float e2 = esq[n];
#pragma unroll
        for (int j = 0; j < 8; ++j) {
            float dot = accHH[j] + (accHL[j] + accLH[j]);
            float d = e2 - 2.0f * dot;
            if (d < minv[j]) { minv[j] = d; mini[j] = n; }   // strict < keeps lowest idx
        }

        wait_asynccnt0();   // next tile's async copies have had the whole compute
        __syncthreads();    // publish next buffer / retire reads of current
    }

    // ---- per-row argmin: butterfly over the 16 lanes holding the same row ----
#pragma unroll
    for (int off = 1; off <= 8; off <<= 1) {
#pragma unroll
        for (int j = 0; j < 8; ++j) {
            float ov = __shfl_xor(minv[j], off, 32);
            int   oi = __shfl_xor(mini[j], off, 32);
            if (ov < minv[j] || (ov == minv[j] && oi < mini[j])) {
                minv[j] = ov; mini[j] = oi;
            }
        }
    }
    if ((lane & 15) == 0) {
        const int rb = wave * 16 + h * 8;   // lane0 -> rows j, lane16 -> rows j+8
#pragma unroll
        for (int j = 0; j < 8; ++j) {
            sIdx[rb + j] = mini[j];
            out_idx[rowBase + rb + j] = (float)mini[j];
        }
    }
    __syncthreads();

    // ---- exact f32 gather of z_q + loss partial ----
    float lsum = 0.f;
    const int r16 = tid >> 4;            // 0..15
    const int c0  = (tid & 15) * 16;     // 16 consecutive floats per thread
    for (int g = 0; g < 8; ++g) {
        const int r = g * 16 + r16;      // 0..127
        const int idx = sIdx[r];
        const float4* wr = (const float4*)(W + (size_t)idx * DIM + c0);
        const float4* zr = (const float4*)(z + (size_t)(rowBase + r) * DIM + c0);
        float4* orow = (float4*)(out_zq + (size_t)(rowBase + r) * DIM + c0);
#pragma unroll
        for (int q = 0; q < 4; ++q) {
            float4 wv = wr[q];
            float4 zv = zr[q];
            orow[q] = wv;                // z_q_ste == z_q numerically
            float dx = wv.x - zv.x, dy = wv.y - zv.y;
            float dz2 = wv.z - zv.z, dw = wv.w - zv.w;
            lsum += dx * dx + dy * dy + dz2 * dz2 + dw * dw;
        }
    }
    sRed[tid] = lsum;
    __syncthreads();
    for (int st = 128; st > 0; st >>= 1) {
        if (tid < st) sRed[tid] += sRed[tid + st];
        __syncthreads();
    }
    if (tid == 0) partials[blockIdx.x] = sRed[0];
}

// loss = (1 + 0.25) * sum((z_q - z)^2) / (N*D), deterministic final reduce
__global__ __launch_bounds__(256) void vq_loss(const float* __restrict__ partials,
                                               float* __restrict__ out_loss) {
    __shared__ float sred[256];
    float v = partials[threadIdx.x] + partials[threadIdx.x + 256];
    sred[threadIdx.x] = v;
    __syncthreads();
    for (int st = 128; st > 0; st >>= 1) {
        if ((int)threadIdx.x < st) sred[threadIdx.x] += sred[threadIdx.x + st];
        __syncthreads();
    }
    if (threadIdx.x == 0)
        out_loss[0] = sred[0] * (1.25f / ((float)N_ROWS * (float)DIM));
}

extern "C" void kernel_launch(void* const* d_in, const int* in_sizes, int n_in,
                              void* d_out, int out_size, void* d_ws, size_t ws_size,
                              hipStream_t stream) {
    (void)in_sizes; (void)n_in; (void)out_size; (void)ws_size;
    const float* z = (const float*)d_in[0];   // [16,4096,256] f32
    const float* W = (const float*)d_in[1];   // [8192,256]   f32

    float* out = (float*)d_out;
    float* out_zq   = out;                                   // 16,777,216 f32
    float* out_idx  = out + (size_t)N_ROWS * DIM;            // 65,536 (argmin as f32)
    float* out_loss = out + (size_t)N_ROWS * DIM + N_ROWS;   // 1

    char* ws = (char*)d_ws;
    bf16_t* whi = (bf16_t*)ws;                               // 4 MB
    bf16_t* wlo = (bf16_t*)(ws + (4u << 20));                // 4 MB
    float*  esq = (float*)(ws + (8u << 20));                 // 32 KB
    float*  partials = (float*)(ws + (8u << 20) + 32768u);   // 2 KB

    vq_prep_codes<<<KCODES, 256, 0, stream>>>(W, whi, wlo);  // 2M elems
    vq_esq<<<KCODES, 256, 0, stream>>>(W, esq);
    vq_main<<<NBLOCKS, 256, 0, stream>>>(z, W, whi, wlo, esq,
                                         out_zq, out_idx, partials);
    vq_loss<<<1, 256, 0, stream>>>(partials, out_loss);
}
